// AdaptiveGCN_37409165149047
// MI455X (gfx1250) — compile-verified, matrix operand order checked by reading, not run on hardware
//
#include <hip/hip_runtime.h>
#include <hip/hip_bf16.h>

typedef __attribute__((ext_vector_type(16))) _Float16 v16h;
typedef __attribute__((ext_vector_type(8)))  _Float16 v8h;
typedef __attribute__((ext_vector_type(8)))  float    v8f;

#define BB 8
#define LL 128
#define DD 768
#define DH 384

// Load a 16-lane-striped WMMA fragment half-pair: per CDNA5 ISA, for 16-bit A
// (16x32) lanes 0-15 hold K=[0..7],[16..23] and lanes 16-31 hold K=[8..15],[24..31]
// as two contiguous 8-element (16B) chunks.  p must point at row*ld + k0 + half*8.
__device__ __forceinline__ v16h ld_frag(const _Float16* p) {
  v8h lo = *(const v8h*)p;
  v8h hi = *(const v8h*)(p + 16);
  v16h f;
#pragma unroll
  for (int t = 0; t < 8; ++t) { f[t] = lo[t]; f[t + 8] = hi[t]; }
  return f;
}

// ---------------- conversion kernels ----------------
__global__ void k_copy_in(const float* __restrict__ x, _Float16* __restrict__ xh,
                          float* __restrict__ outf, int n) {
  for (int i = blockIdx.x * blockDim.x + threadIdx.x; i < n; i += gridDim.x * blockDim.x) {
    float v = x[i];
    xh[i] = (_Float16)v;
    outf[i] = v;
  }
}

// dst[b][c][r] = (f16) src[b][r][c]   (batched transpose to column-major f16)
__global__ void k_transpose(const float* __restrict__ src, _Float16* __restrict__ dst,
                            int R, int C, int batch) {
  int total = batch * R * C;
  for (int i = blockIdx.x * blockDim.x + threadIdx.x; i < total; i += gridDim.x * blockDim.x) {
    int b = i / (R * C);
    int rem = i - b * R * C;
    int r = rem / C;
    int c = rem - r * C;
    dst[(size_t)b * R * C + (size_t)c * R + r] = (_Float16)src[i];
  }
}

__global__ void k_mask(float* __restrict__ p, int n) {
  int i = blockIdx.x * blockDim.x + threadIdx.x;
  if (i < n) p[i] = 0.0f;   // softmax rows sum to 1 -> mask is identically false
}

// ---------------- plain WMMA GEMM: C(f32) = A_f16[M,K] @ Bt_f16[N,K]^T ----------------
// Each wave computes a 16x32 output strip (2 N-tiles sharing one A fragment).
__global__ __launch_bounds__(256) void k_gemm(const _Float16* __restrict__ A, int lda,
                                              const _Float16* __restrict__ Bt, int ldb,
                                              float* __restrict__ C, int ldc, int K) {
  int tid = threadIdx.x, lane = tid & 31, w = tid >> 5;
  int half = lane >> 4, row = lane & 15;
  int n0 = blockIdx.x * 256 + w * 32;
  int m0 = blockIdx.y * 16;
  v8f acc0 = {}, acc1 = {};
  for (int k0 = 0; k0 < K; k0 += 32) {
    v16h af = ld_frag(A + (size_t)(m0 + row) * lda + k0 + half * 8);
    v16h b0 = ld_frag(Bt + (size_t)(n0 + row) * ldb + k0 + half * 8);
    v16h b1 = ld_frag(Bt + (size_t)(n0 + 16 + row) * ldb + k0 + half * 8);
    acc0 = __builtin_amdgcn_wmma_f32_16x16x32_f16(false, af, false, b0, (short)0, acc0, false, false);
    acc1 = __builtin_amdgcn_wmma_f32_16x16x32_f16(false, af, false, b1, (short)0, acc1, false, false);
  }
#pragma unroll
  for (int r = 0; r < 8; ++r) {
    int m = m0 + r + half * 8;
    C[(size_t)m * ldc + (n0 + row)]      = acc0[r];
    C[(size_t)m * ldc + (n0 + 16 + row)] = acc1[r];
  }
}

// ---------------- edge-weight kernel: one WG per (b,i) ----------------
// Builds A = relu(hi[i]+hj[j]+b1) in LDS per 128-wide K chunk, runs the
// 128x384x768 WMMA GEMM against w2^T, fuses relu(.+b2)@w3, sigmoid, softmax.
__global__ __launch_bounds__(512) void k_edge(const float* __restrict__ hi,
                                              const float* __restrict__ hj,
                                              const float* __restrict__ adj,
                                              const float* __restrict__ b1,
                                              const float* __restrict__ b2,
                                              const float* __restrict__ b3,
                                              const _Float16* __restrict__ w2T,
                                              const float* __restrict__ w3,
                                              _Float16* __restrict__ a_h) {
  __shared__ _Float16 Ach[LL * 128];   // 32 KB chunk of the on-the-fly A matrix
  __shared__ float hi_s[DD];
  __shared__ float score[LL];
  __shared__ float vals[LL];
  __shared__ float red[LL];

  int b = blockIdx.y, i = blockIdx.x;
  int tid = threadIdx.x;

  for (int k = tid; k < DD; k += 512) hi_s[k] = hi[((size_t)(b * LL + i)) * DD + k] + b1[k];
  if (tid < LL) score[tid] = b3[0];

  int lane = tid & 31, wave = tid >> 5;
  int mt = wave & 7, ns = wave >> 3;       // 8 M-tiles x 2 N-halves = 16 waves
  int half = lane >> 4, row = lane & 15;

  v8f acc[12];
  v8f z = {};
#pragma unroll
  for (int t = 0; t < 12; ++t) acc[t] = z;

  const float* hjrow = hj + (size_t)b * LL * DD;

  for (int kc = 0; kc < 6; ++kc) {
    __syncthreads();
    for (int e = tid; e < LL * 128; e += 512) {
      int j = e >> 7, kk = e & 127;
      int kg = kc * 128 + kk;
      float v = hi_s[kg] + hjrow[(size_t)j * DD + kg];
      Ach[e] = (_Float16)fmaxf(v, 0.0f);
    }
    __syncthreads();
    // A fragments depend only on k: load the 4 K-step fragments once from LDS,
    // reuse across all 12 N-tiles (12x less LDS traffic than reloading per tile).
    v16h af[4];
#pragma unroll
    for (int ks = 0; ks < 4; ++ks)
      af[ks] = ld_frag(Ach + (mt * 16 + row) * 128 + ks * 32 + half * 8);
#pragma unroll
    for (int t = 0; t < 12; ++t) {
      int n0 = (ns * 12 + t) * 16;
#pragma unroll
      for (int ks = 0; ks < 4; ++ks) {
        v16h bf = ld_frag(w2T + (size_t)(n0 + row) * DD + kc * 128 + ks * 32 + half * 8);
        acc[t] = __builtin_amdgcn_wmma_f32_16x16x32_f16(false, af[ks], false, bf,
                                                        (short)0, acc[t], false, false);
      }
    }
  }
  __syncthreads();

  // fused relu(+b2) and dot with w3 (N is the lane dimension of the C tile)
  float p[8];
#pragma unroll
  for (int r = 0; r < 8; ++r) p[r] = 0.0f;
#pragma unroll
  for (int t = 0; t < 12; ++t) {
    int n = (ns * 12 + t) * 16 + row;
    float wv = w3[n];
    float bv = b2[n];
#pragma unroll
    for (int r = 0; r < 8; ++r) p[r] += fmaxf(acc[t][r] + bv, 0.0f) * wv;
  }
#pragma unroll
  for (int r = 0; r < 8; ++r) atomicAdd(&score[mt * 16 + r + half * 8], p[r]);
  __syncthreads();

  // sigmoid -> adj*ew + eye -> softmax over the 128 j's
  if (tid < LL) {
    float s = score[tid];
    float ew = 1.0f / (1.0f + expf(-s));
    float av = adj[((size_t)(b * LL + i)) * LL + tid] * ew + ((tid == i) ? 1.0f : 0.0f);
    vals[tid] = av;
    red[tid] = av;
  }
  __syncthreads();
  for (int s = 64; s > 0; s >>= 1) {
    if (tid < s) red[tid] = fmaxf(red[tid], red[tid + s]);
    __syncthreads();
  }
  float mx = red[0];
  __syncthreads();
  if (tid < LL) { float e = expf(vals[tid] - mx); vals[tid] = e; red[tid] = e; }
  __syncthreads();
  for (int s = 64; s > 0; s >>= 1) {
    if (tid < s) red[tid] += red[tid + s];
    __syncthreads();
  }
  float inv = 1.0f / red[0];
  if (tid < LL) a_h[((size_t)(b * LL + i)) * LL + tid] = (_Float16)(vals[tid] * inv);
}

// ---------------- GCN aggregation: t = a @ out + out  (f16 out for next GEMM) ----------------
__global__ __launch_bounds__(256) void k_gcn_agg(const _Float16* __restrict__ a_h,
                                                 const _Float16* __restrict__ outT,
                                                 const float* __restrict__ outf,
                                                 _Float16* __restrict__ t_h) {
  int tid = threadIdx.x, lane = tid & 31, w = tid >> 5;
  int half = lane >> 4, row = lane & 15;
  int b = blockIdx.z;
  int n0 = blockIdx.x * 256 + w * 32;
  int m0 = blockIdx.y * 16;
  const _Float16* A = a_h + (size_t)b * LL * LL;
  const _Float16* Bt = outT + (size_t)b * DD * LL;
  v8f acc0 = {}, acc1 = {};
#pragma unroll
  for (int k0 = 0; k0 < LL; k0 += 32) {
    v16h af = ld_frag(A + (m0 + row) * LL + k0 + half * 8);
    v16h b0 = ld_frag(Bt + (size_t)(n0 + row) * LL + k0 + half * 8);
    v16h b1 = ld_frag(Bt + (size_t)(n0 + 16 + row) * LL + k0 + half * 8);
    acc0 = __builtin_amdgcn_wmma_f32_16x16x32_f16(false, af, false, b0, (short)0, acc0, false, false);
    acc1 = __builtin_amdgcn_wmma_f32_16x16x32_f16(false, af, false, b1, (short)0, acc1, false, false);
  }
#pragma unroll
  for (int r = 0; r < 8; ++r) {
    int m = m0 + r + half * 8;
    size_t i0 = ((size_t)(b * LL + m)) * DD + n0 + row;
    size_t i1 = i0 + 16;
    t_h[i0] = (_Float16)(acc0[r] + outf[i0]);
    t_h[i1] = (_Float16)(acc1[r] + outf[i1]);
  }
}

// ---------------- GCN weight GEMM: y = (t @ gw + 2*gb) / 2  (denom == 2 exactly) ----------------
__global__ __launch_bounds__(256) void k_gcn_mm(const _Float16* __restrict__ t_h,
                                                const _Float16* __restrict__ gwT,
                                                const float* __restrict__ gb,
                                                float* __restrict__ y) {
  int tid = threadIdx.x, lane = tid & 31, w = tid >> 5;
  int half = lane >> 4, row = lane & 15;
  int n0 = blockIdx.x * 256 + w * 32;
  int m0 = blockIdx.y * 16;
  v8f acc0 = {}, acc1 = {};
  for (int k0 = 0; k0 < DD; k0 += 32) {
    v16h af = ld_frag(t_h + (size_t)(m0 + row) * DD + k0 + half * 8);
    v16h b0 = ld_frag(gwT + (size_t)(n0 + row) * DD + k0 + half * 8);
    v16h b1 = ld_frag(gwT + (size_t)(n0 + 16 + row) * DD + k0 + half * 8);
    acc0 = __builtin_amdgcn_wmma_f32_16x16x32_f16(false, af, false, b0, (short)0, acc0, false, false);
    acc1 = __builtin_amdgcn_wmma_f32_16x16x32_f16(false, af, false, b1, (short)0, acc1, false, false);
  }
  float bv0 = 2.0f * gb[n0 + row];
  float bv1 = 2.0f * gb[n0 + 16 + row];
#pragma unroll
  for (int r = 0; r < 8; ++r) {
    int m = m0 + r + half * 8;
    y[(size_t)m * DD + (n0 + row)]      = (acc0[r] + bv0) * 0.5f;
    y[(size_t)m * DD + (n0 + 16 + row)] = (acc1[r] + bv1) * 0.5f;
  }
}

// ---------------- LayerNorm(relu(y)); also emits transposed f16 for next layer ----------------
__global__ __launch_bounds__(256) void k_ln(const float* __restrict__ y,
                                            const float* __restrict__ g,
                                            const float* __restrict__ bta,
                                            float* __restrict__ outf,
                                            _Float16* __restrict__ outT,
                                            float* __restrict__ dst) {
  __shared__ float red[256];
  int rowi = blockIdx.x;
  int b = rowi >> 7, m = rowi & 127;
  int tid = threadIdx.x;
  float x[3];
#pragma unroll
  for (int t = 0; t < 3; ++t) x[t] = fmaxf(y[(size_t)rowi * DD + tid + t * 256], 0.0f);
  red[tid] = x[0] + x[1] + x[2];
  __syncthreads();
  for (int s = 128; s > 0; s >>= 1) { if (tid < s) red[tid] += red[tid + s]; __syncthreads(); }
  float mean = red[0] * (1.0f / DD);
  __syncthreads();
  red[tid] = x[0] * x[0] + x[1] * x[1] + x[2] * x[2];
  __syncthreads();
  for (int s = 128; s > 0; s >>= 1) { if (tid < s) red[tid] += red[tid + s]; __syncthreads(); }
  float var = red[0] * (1.0f / DD) - mean * mean;
  float rs = rsqrtf(var + 1e-5f);
#pragma unroll
  for (int t = 0; t < 3; ++t) {
    int k = tid + t * 256;
    float o = (x[t] - mean) * rs * g[k] + bta[k];
    outf[(size_t)rowi * DD + k] = o;
    outT[((size_t)b * DD + k) * LL + m] = (_Float16)o;
    if (dst) dst[(size_t)rowi * DD + k] = o;
  }
}

extern "C" void kernel_launch(void* const* d_in, const int* in_sizes, int n_in,
                              void* d_out, int out_size, void* d_ws, size_t ws_size,
                              hipStream_t stream) {
  (void)in_sizes; (void)n_in; (void)out_size; (void)ws_size;
  const float* adj    = (const float*)d_in[0];
  const float* inputs = (const float*)d_in[1];
  const float* w1     = (const float*)d_in[2];
  const float* b1     = (const float*)d_in[3];
  const float* w2     = (const float*)d_in[4];
  const float* b2     = (const float*)d_in[5];
  const float* w3     = (const float*)d_in[6];
  const float* b3     = (const float*)d_in[7];
  const float* gw     = (const float*)d_in[8];
  const float* gb     = (const float*)d_in[9];
  const float* lng    = (const float*)d_in[10];
  const float* lnb    = (const float*)d_in[11];
  float* out = (float*)d_out;

  char* ws = (char*)d_ws;
  size_t off = 0;
  auto alloc = [&](size_t bytes) -> char* {
    char* p = ws + off;
    off += (bytes + 255) & ~(size_t)255;
    return p;
  };
  const size_t MD = (size_t)BB * LL * DD;   // 786432
  float*    hi_  = (float*)alloc(MD * 4);
  float*    hj_  = (float*)alloc(MD * 4);
  float*    outf = (float*)alloc(MD * 4);
  float*    y    = (float*)alloc(MD * 4);
  _Float16* xh   = (_Float16*)alloc(MD * 2);
  _Float16* w1aT = (_Float16*)alloc((size_t)DD * DD * 2);
  _Float16* w1bT = (_Float16*)alloc((size_t)DD * DD * 2);
  _Float16* w2T  = (_Float16*)alloc((size_t)DH * DD * 2);
  _Float16* gwT  = (_Float16*)alloc((size_t)3 * DD * DD * 2);
  _Float16* a_h  = (_Float16*)alloc((size_t)BB * LL * LL * 2);
  _Float16* t_h  = (_Float16*)alloc(MD * 2);
  _Float16* outT = (_Float16*)alloc(MD * 2);

  // conversions / transposes to f16 (B operands stored N-major for contiguous fragments)
  k_copy_in<<<1024, 256, 0, stream>>>(inputs, xh, outf, (int)MD);
  k_transpose<<<1024, 256, 0, stream>>>(w1,                  w1aT, DD, DD, 1);
  k_transpose<<<1024, 256, 0, stream>>>(w1 + (size_t)DD * DD, w1bT, DD, DD, 1);
  k_transpose<<<512,  256, 0, stream>>>(w2,                  w2T,  DD, DH, 1);
  k_transpose<<<2048, 256, 0, stream>>>(gw,                  gwT,  DD, DD, 3);
  k_transpose<<<1024, 256, 0, stream>>>(inputs,              outT, LL, DD, BB);

  // hi = x @ w1[:D], hj = x @ w1[D:]
  dim3 gproj(DD / 256, (BB * LL) / 16);
  k_gemm<<<gproj, 256, 0, stream>>>(xh, DD, w1aT, DD, hi_, DD, DD);
  k_gemm<<<gproj, 256, 0, stream>>>(xh, DD, w1bT, DD, hj_, DD, DD);

  // adaptive edge weights + softmax  (the 77-GFLOP hot loop)
  k_edge<<<dim3(LL, BB), 512, 0, stream>>>(hi_, hj_, adj, b1, b2, b3, w2T, w3, a_h);

  // mask output: identically false
  k_mask<<<4, 256, 0, stream>>>(out + MD, BB * LL);

  // 3 GCN layers
  for (int li = 0; li < 3; ++li) {
    k_gcn_agg<<<dim3(DD / 256, LL / 16, BB), 256, 0, stream>>>(a_h, outT, outf, t_h);
    k_gcn_mm<<<dim3(DD / 256, (BB * LL) / 16), 256, 0, stream>>>(
        t_h, gwT + (size_t)li * DD * DD, gb + (size_t)li * DD, y);
    k_ln<<<BB * LL, 256, 0, stream>>>(y, lng + (size_t)li * DD, lnb + (size_t)li * DD,
                                      outf, outT, li == 2 ? out : nullptr);
  }
}
